// GraphGRUCell_33011118637673
// MI455X (gfx1250) — matrix operand order repaired and checked by reading, not compiled, not run on hardware
//
#include <hip/hip_runtime.h>
#include <math.h>

// ---------------------------------------------------------------------------
// GraphGRUCell for MI455X (gfx1250, wave32).
//   B=8, N=10000, U=64, D=1, E=320000.
// Pipeline:
//   1) CSR build for dst-grouped edges (+self loops): count / scan / fill
//   2) GEMM1 (WMMA f16-split): h = [state|inp] @ W_gat ; fused a_s/a_d dots
//   3) Aggregate: 1 wave per (b,dst), online softmax over edge list, reg accum
//   4) GRU (WMMA f16-split): sigmoid gates, r*sg via LDS relayout, tanh, blend
// WMMA uses V_WMMA_F32_16X16X32_F16 with hi/lo f16 splitting:
//   x = hi + lo  =>  A@B ~= Ahi@Bhi + Alo@Bhi + Ahi@Blo   (f32 accumulate)
// ---------------------------------------------------------------------------

typedef __attribute__((ext_vector_type(16))) _Float16 v16h;
typedef __attribute__((ext_vector_type(8)))  float    v8f;

#define WMMA_F16(A, B, C) \
  __builtin_amdgcn_wmma_f32_16x16x32_f16(false, (A), false, (B), (short)0, (C), false, false)

// ---------------------------------------------------------------------------
// Weight pre-pack into LDS, laid out exactly as the per-lane B operand:
//   pack[(((ct*2+kb)*2+part)*32 + lane)*16 + j]
//   lane<16 : col = ct*16+lane, K = kb*32 + j (j<8), kb*32 + 16 + j-8 (j>=8)
//   lane>=16: col = ct*16+lane-16, K = kb*32 + 8 + j (j<8), kb*32 + 24 + j-8 (j>=8)
//   part 0 = f16(hi), part 1 = f16(residual)
// krow_off lets us skip the input row (concat order differs per matmul).
// ---------------------------------------------------------------------------
__device__ inline void prepack_weights(const float* __restrict__ W, int ldw,
                                       int ct_count, int krow_off,
                                       _Float16* pack) {
  const int total = ct_count * 2048;  // ct*2kb*2part*32lane*16j
  for (int idx = threadIdx.x; idx < total; idx += blockDim.x) {
    const int j    = idx & 15;
    const int lane = (idx >> 4) & 31;
    const int part = (idx >> 9) & 1;
    const int kb   = (idx >> 10) & 1;
    const int ct   = idx >> 11;
    const int col  = ct * 16 + (lane & 15);
    const int krow = kb * 32 + ((j >> 3) << 4) + ((lane >> 4) << 3) + (j & 7);
    const float w  = W[(krow + krow_off) * ldw + col];
    const _Float16 hi = (_Float16)w;
    pack[idx] = part ? (_Float16)(w - (float)hi) : hi;
  }
}

// ---------------------------------------------------------------------------
// 16-row tile GEMM, K = 64, CT column tiles of 16.  A rows are f32 (lda
// stride), B pre-packed hi/lo in LDS.  acc[] is the C/D f32 layout:
//   acc[ct][i] -> row (i + 8*(lane>=16)), col (ct*16 + (lane&15))
// A operand layout (16-bit A 16x32): lane<16 holds row=lane,
//   elements 0..7 -> K 0..7, elements 8..15 -> K 16..23;
//   lane>=16 holds row=lane-16, K 8..15 / 24..31.
// ---------------------------------------------------------------------------
template <int CT>
__device__ inline void wmma_rowtile_k64(const float* __restrict__ A, int lda,
                                        const _Float16* __restrict__ pack,
                                        v8f* acc) {
  const int lane   = threadIdx.x & 31;
  const int halfid = lane >> 4;
  const int r      = lane & 15;
#pragma unroll
  for (int kb = 0; kb < 2; ++kb) {
    v16h ahi, alo;
    const float* s0 = A + r * lda + kb * 32 + 8 * halfid;
#pragma unroll
    for (int j = 0; j < 8; ++j) {
      float x = s0[j];
      _Float16 hx = (_Float16)x;
      ahi[j] = hx;
      alo[j] = (_Float16)(x - (float)hx);
      float y = s0[16 + j];
      _Float16 hy = (_Float16)y;
      ahi[8 + j] = hy;
      alo[8 + j] = (_Float16)(y - (float)hy);
    }
#pragma unroll
    for (int ct = 0; ct < CT; ++ct) {
      const _Float16* bb = pack + ((ct * 2 + kb) * 2) * 512 + lane * 16;
      v16h bhi = *(const v16h*)bb;
      v16h blo = *(const v16h*)(bb + 512);
      acc[ct] = WMMA_F16(ahi, bhi, acc[ct]);
      acc[ct] = WMMA_F16(alo, bhi, acc[ct]);
      acc[ct] = WMMA_F16(ahi, blo, acc[ct]);
    }
  }
}

// ---------------------------------------------------------------------------
// CSR build kernels
// ---------------------------------------------------------------------------
__global__ void init_cnt_kernel(int* cnt, int N) {
  int i = blockIdx.x * blockDim.x + threadIdx.x;
  if (i < N) cnt[i] = 1;  // 1 = the self loop
}

__global__ void count_kernel(const int* __restrict__ edges, int E, int* cnt) {
  int e = blockIdx.x * blockDim.x + threadIdx.x;
  if (e < E) atomicAdd(&cnt[edges[E + e]], 1);  // row 1 = dst
}

// Single-block exclusive scan over N<=10240 counts -> off[0..N], cur copy.
__global__ void __launch_bounds__(1024) scan_kernel(const int* __restrict__ cnt,
                                                    int N, int* off, int* cur) {
  __shared__ int sh[1024];
  const int t = threadIdx.x;
  const int CH = 10;  // ceil(10000/1024)
  int loc[CH];
  int base = t * CH, sum = 0;
#pragma unroll
  for (int i = 0; i < CH; ++i) {
    int v = (base + i < N) ? cnt[base + i] : 0;
    loc[i] = sum;
    sum += v;
  }
  sh[t] = sum;
  __syncthreads();
  for (int o = 1; o < 1024; o <<= 1) {
    int v = (t >= o) ? sh[t - o] : 0;
    __syncthreads();
    sh[t] += v;
    __syncthreads();
  }
  int prefix = (t == 0) ? 0 : sh[t - 1];
#pragma unroll
  for (int i = 0; i < CH; ++i) {
    int idx = base + i;
    if (idx < N) {
      int o = prefix + loc[i];
      off[idx] = o;
      cur[idx] = o;
    }
  }
  if (t == 1023) off[N] = sh[1023];
}

__global__ void fill_kernel(const int* __restrict__ edges, int E, int N,
                            int* cur, int* srcList) {
  int e = blockIdx.x * blockDim.x + threadIdx.x;
  if (e >= E + N) return;
  int s, d;
  if (e < E) { s = edges[e]; d = edges[E + e]; }
  else       { s = d = e - E; }
  int pos = atomicAdd(&cur[d], 1);
  srcList[pos] = s;
}

// ---------------------------------------------------------------------------
// GEMM1: h = [state | inputs] @ W_gat, fused attention dots a_s/a_d.
// W_gat rows 0..63 -> state, row 64 -> the single input column (rank-1).
// ---------------------------------------------------------------------------
#define G1_WAVES 4
__global__ void __launch_bounds__(128)
gemm1_kernel(const float* __restrict__ state, const float* __restrict__ inputs,
             const float* __restrict__ Wgat, const float* __restrict__ att_src,
             const float* __restrict__ att_dst, float* __restrict__ h,
             float* __restrict__ a_s, float* __restrict__ a_d) {
  __shared__ __align__(32) _Float16 wpack[4 * 2048];  // 16 KB
  prepack_weights(Wgat, 64, 4, /*krow_off=*/0, wpack);
  __syncthreads();

  const int lane = threadIdx.x & 31, halfid = lane >> 4, r = lane & 15;
  const int tile = blockIdx.x * G1_WAVES + (threadIdx.x >> 5);
  const int m0 = tile * 16;

  v8f acc[4] = {};
  wmma_rowtile_k64<4>(state + m0 * 64, 64, wpack, acc);

  float inp[8];
#pragma unroll
  for (int i = 0; i < 8; ++i) inp[i] = inputs[m0 + i + 8 * halfid];

  float ps[8] = {}, pd[8] = {};
#pragma unroll
  for (int ct = 0; ct < 4; ++ct) {
    const int col = ct * 16 + r;
    const float wtop = Wgat[64 * 64 + col];  // input row of W_gat
    const float asc = att_src[col], adc = att_dst[col];
#pragma unroll
    for (int i = 0; i < 8; ++i) {
      const int row = i + 8 * halfid;
      float v = acc[ct][i] + inp[i] * wtop;
      h[(m0 + row) * 64 + col] = v;
      ps[i] += v * asc;
      pd[i] += v * adc;
    }
  }
  // Reduce across the 16 lanes of each half-wave (rows differ per half).
#pragma unroll
  for (int o = 8; o >= 1; o >>= 1) {
#pragma unroll
    for (int i = 0; i < 8; ++i) {
      ps[i] += __shfl_xor(ps[i], o, 32);
      pd[i] += __shfl_xor(pd[i], o, 32);
    }
  }
  if (r == 0) {
#pragma unroll
    for (int i = 0; i < 8; ++i) {
      a_s[m0 + i + 8 * halfid] = ps[i];
      a_d[m0 + i + 8 * halfid] = pd[i];
    }
  }
}

// ---------------------------------------------------------------------------
// Aggregation: one wave per (b, dst). Online softmax over the dst's edge
// list; 64-float accumulator held 2 floats per lane. Adds b_gat.
// ---------------------------------------------------------------------------
__global__ void __launch_bounds__(256)
agg_kernel(const float* __restrict__ h, const float* __restrict__ a_s,
           const float* __restrict__ a_d, const float* __restrict__ b_gat,
           const int* __restrict__ off, const int* __restrict__ srcList,
           float* __restrict__ sg, int N, int BN) {
  const int wid = (blockIdx.x * blockDim.x + threadIdx.x) >> 5;
  const int lane = threadIdx.x & 31;
  if (wid >= BN) return;
  const int n = wid % N;
  const int bN = wid - n;  // b*N

  const float ad = a_d[wid];
  const int s = off[n], e = off[n + 1];
  float m = -INFINITY, ssum = 0.f, acc0 = 0.f, acc1 = 0.f;
  for (int k = s; k < e; ++k) {
    const int src = srcList[k];
    if (k + 4 < e) __builtin_prefetch(&srcList[k + 4], 0, 0);
    float logit = a_s[bN + src] + ad;
    logit = (logit > 0.f) ? logit : 0.2f * logit;  // leaky_relu(0.2)
    const float mn = fmaxf(m, logit);
    const float scale = expf(m - mn);  // first iter: exp(-inf)=0
    const float w = expf(logit - mn);
    ssum = ssum * scale + w;
    const float* hr = h + (bN + src) * 64;
    acc0 = acc0 * scale + w * hr[lane];
    acc1 = acc1 * scale + w * hr[lane + 32];
    m = mn;
  }
  const float inv = 1.f / ssum;
  sg[wid * 64 + lane]      = acc0 * inv + b_gat[lane];
  sg[wid * 64 + lane + 32] = acc1 * inv + b_gat[lane + 32];
}

// ---------------------------------------------------------------------------
// GRU: value = sigmoid([inp|sg]@W1+b1) -> r,u ; c = tanh([inp|r*sg]@W2+b2)
// out = u*sg + (1-u)*c.   W1/W2 row 0 = input column, rows 1..64 = hidden.
// ---------------------------------------------------------------------------
#define GRU_WAVES 4
__global__ void __launch_bounds__(128)
gru_kernel(const float* __restrict__ inputs, const float* __restrict__ sg,
           const float* __restrict__ W1, const float* __restrict__ b1,
           const float* __restrict__ W2, const float* __restrict__ b2,
           float* __restrict__ out) {
  __shared__ __align__(32) _Float16 w1pack[8 * 2048];  // 32 KB
  __shared__ __align__(32) _Float16 w2pack[4 * 2048];  // 16 KB
  __shared__ float stash[GRU_WAVES][2][16 * 64];       // [0]=r*sg, [1]=u ; 32 KB
  prepack_weights(W1, 128, 8, /*krow_off=*/1, w1pack);
  prepack_weights(W2, 64, 4, /*krow_off=*/1, w2pack);
  __syncthreads();

  const int wslot = threadIdx.x >> 5;
  const int lane = threadIdx.x & 31, halfid = lane >> 4, r = lane & 15;
  const int tile = blockIdx.x * GRU_WAVES + wslot;
  const int m0 = tile * 16;

  v8f acc1[8] = {};
  wmma_rowtile_k64<8>(sg + m0 * 64, 64, w1pack, acc1);

  float inp[8];
#pragma unroll
  for (int i = 0; i < 8; ++i) inp[i] = inputs[m0 + i + 8 * halfid];

#pragma unroll
  for (int ct = 0; ct < 8; ++ct) {
    const int col = ct * 16 + r;
    const float wtop = W1[col];  // input row (row 0) of W1
    const float bias = b1[col];
#pragma unroll
    for (int i = 0; i < 8; ++i) {
      const int row = i + 8 * halfid;
      float v = acc1[ct][i] + inp[i] * wtop + bias;
      v = 1.f / (1.f + expf(-v));  // sigmoid
      if (ct < 4) {                // r-gate (cols 0..63): stage r*sg for GEMM2
        stash[wslot][0][row * 64 + col] = v * sg[(m0 + row) * 64 + col];
      } else {                     // u-gate (cols 64..127)
        stash[wslot][1][row * 64 + (col - 64)] = v;
      }
    }
  }
  __syncthreads();  // all waves of this exact-sized block reach here

  v8f acc2[4] = {};
  wmma_rowtile_k64<4>(&stash[wslot][0][0], 64, w2pack, acc2);

#pragma unroll
  for (int ct = 0; ct < 4; ++ct) {
    const int col = ct * 16 + r;
    const float wtop = W2[col];  // input row (row 0) of W2
    const float bias = b2[col];
#pragma unroll
    for (int i = 0; i < 8; ++i) {
      const int row = i + 8 * halfid;
      const float c = tanhf(acc2[ct][i] + inp[i] * wtop + bias);
      const float u = stash[wslot][1][row * 64 + col];
      const float sgv = sg[(m0 + row) * 64 + col];
      out[(m0 + row) * 64 + col] = u * sgv + (1.f - u) * c;
    }
  }
}

// ---------------------------------------------------------------------------
// Host launcher
// ---------------------------------------------------------------------------
extern "C" void kernel_launch(void* const* d_in, const int* in_sizes, int n_in,
                              void* d_out, int out_size, void* d_ws,
                              size_t ws_size, hipStream_t stream) {
  (void)in_sizes; (void)n_in; (void)out_size; (void)ws_size;
  constexpr int B = 8, N = 10000, E = 320000;
  constexpr int BN = B * N;

  const float* inputs  = (const float*)d_in[0];   // [B*N]
  const float* state   = (const float*)d_in[1];   // [B*N, 64]
  const int*   edges   = (const int*)d_in[2];     // [2, E]
  const float* W_gat   = (const float*)d_in[3];   // [65, 64]
  const float* att_src = (const float*)d_in[4];   // [64]
  const float* att_dst = (const float*)d_in[5];   // [64]
  const float* b_gat   = (const float*)d_in[6];   // [64]
  const float* W1      = (const float*)d_in[7];   // [65, 128]
  const float* b1      = (const float*)d_in[8];   // [128]
  const float* W2      = (const float*)d_in[9];   // [65, 64]
  const float* b2      = (const float*)d_in[10];  // [64]
  float* out = (float*)d_out;                     // [B*N, 64]

  // Workspace carve-out (~43 MB total)
  char* ws = (char*)d_ws;
  auto carve = [&](size_t bytes) {
    void* p = (void*)ws;
    ws += (bytes + 511) & ~size_t(511);
    return p;
  };
  float* h   = (float*)carve((size_t)BN * 64 * sizeof(float));
  float* sg  = (float*)carve((size_t)BN * 64 * sizeof(float));
  float* a_s = (float*)carve((size_t)BN * sizeof(float));
  float* a_d = (float*)carve((size_t)BN * sizeof(float));
  int* cnt     = (int*)carve((size_t)(N + 1) * sizeof(int));
  int* off     = (int*)carve((size_t)(N + 1) * sizeof(int));
  int* cur     = (int*)carve((size_t)(N + 1) * sizeof(int));
  int* srcList = (int*)carve((size_t)(E + N) * sizeof(int));

  // 1) CSR build (dst-grouped edge lists incl. self loops)
  init_cnt_kernel<<<(N + 255) / 256, 256, 0, stream>>>(cnt, N);
  count_kernel<<<(E + 255) / 256, 256, 0, stream>>>(edges, E, cnt);
  scan_kernel<<<1, 1024, 0, stream>>>(cnt, N, off, cur);
  fill_kernel<<<(E + N + 255) / 256, 256, 0, stream>>>(edges, E, N, cur, srcList);

  // 2) GATConv linear + attention scalars (WMMA)
  gemm1_kernel<<<BN / 16 / G1_WAVES, 128, 0, stream>>>(
      state, inputs, W_gat, att_src, att_dst, h, a_s, a_d);

  // 3) Per-dst online-softmax aggregation (wave per node)
  agg_kernel<<<(BN * 32 + 255) / 256, 256, 0, stream>>>(
      h, a_s, a_d, b_gat, off, srcList, sg, N, BN);

  // 4) GRU gating (WMMA x2, fused)
  gru_kernel<<<BN / 16 / GRU_WAVES, 128, 0, stream>>>(
      inputs, sg, W1, b1, W2, b2, out);
}
